// ImageTransformer_Seq_DETR_MS_49435073577594
// MI455X (gfx1250) — compile-verified, matrix operand order checked by reading, not run on hardware
//
#include <hip/hip_runtime.h>

#define B_   2
#define V_   6
#define C_   128
#define H_   8
#define P_   200
#define N_   7000
#define FFN_ 512
#define BV_  (B_*V_)
#define DH_  16
#define QT_  ((P_ + 15) / 16)
#define SPLIT_ 4

typedef __attribute__((ext_vector_type(16))) __bf16 v16bf;
typedef __attribute__((ext_vector_type(8)))  float  v8f;
typedef unsigned short u16;
typedef unsigned int   u32;

union Frag { v16bf v; u32 u[8]; };

__device__ __forceinline__ u16 f2bf(float f) {
  u32 x = __builtin_bit_cast(u32, f);
  x += 0x7fffu + ((x >> 16) & 1u);
  return (u16)(x >> 16);
}
__device__ __forceinline__ float bf2f(u16 h) {
  u32 x = ((u32)h) << 16;
  return __builtin_bit_cast(float, x);
}

// ---------------------------------------------------------------------------
// Generic bf16 WMMA GEMM:  out[M,Ncol] = A[M,K] * W^T  (W stored [Ncol,K]) + bias
// Each wave computes a 32x64 strip: 2 A fragments x 4 B fragments -> 8 WMMAs
// per K-step, each B fragment reused by two WMMAs (~21 FLOP/byte from L2).
// K multiple of 32; Ncol multiple of 64.
// ---------------------------------------------------------------------------
__global__ void gemm_bf16_kernel(const u16* __restrict__ A, const u16* __restrict__ W,
                                 const float* __restrict__ bias,
                                 const u16* __restrict__ addB,
                                 float* __restrict__ outF, u16* __restrict__ outB,
                                 int M, int Ncol, int K, int relu)
{
  int lane = threadIdx.x & 31;
  int wave = threadIdx.x >> 5;
  int rowTile = (blockIdx.y * 4 + wave) * 32;
  if (rowTile >= M) return;
  int r0 = rowTile + (lane & 15);      if (r0 >= M) r0 = M - 1;
  int r1 = rowTile + 16 + (lane & 15); if (r1 >= M) r1 = M - 1;
  int khalf = (lane >> 4) * 8;
  int col = lane & 15;
  int colBase = blockIdx.x * 64;

  v8f acc0[4], acc1[4];
#pragma unroll
  for (int t = 0; t < 4; ++t) {
    int c = colBase + t * 16 + col; if (c >= Ncol) c = Ncol - 1;
    float bv = bias ? bias[c] : 0.0f;
    acc0[t] = (v8f){bv, bv, bv, bv, bv, bv, bv, bv};
    acc1[t] = acc0[t];
  }
  for (int kb = 0; kb < K; kb += 32) {
    Frag a0, a1;
    const u16* ap0 = A + (size_t)r0 * K + kb;
    const u16* ap1 = A + (size_t)r1 * K + kb;
#pragma unroll
    for (int j = 0; j < 4; ++j) {
      a0.u[j]     = *(const u32*)(ap0 + khalf + 2 * j);
      a0.u[4 + j] = *(const u32*)(ap0 + 16 + khalf + 2 * j);
      a1.u[j]     = *(const u32*)(ap1 + khalf + 2 * j);
      a1.u[4 + j] = *(const u32*)(ap1 + 16 + khalf + 2 * j);
    }
    if (kb + 32 < K) {  // gfx1250 global_prefetch_b8 for next K-chunk of A
      __builtin_prefetch(ap0 + 32, 0, 0);
      __builtin_prefetch(ap1 + 32, 0, 0);
    }
#pragma unroll
    for (int t = 0; t < 4; ++t) {
      int c = colBase + t * 16 + col; if (c >= Ncol) c = Ncol - 1;
      const u16* wp = W + (size_t)c * K + kb;
      Frag b;
#pragma unroll
      for (int j = 0; j < 4; ++j) {
        b.u[j]     = *(const u32*)(wp + khalf + 2 * j);
        b.u[4 + j] = *(const u32*)(wp + 16 + khalf + 2 * j);
      }
      acc0[t] = __builtin_amdgcn_wmma_f32_16x16x32_bf16(false, a0.v, false, b.v,
                                                        (short)0, acc0[t], false, false);
      acc1[t] = __builtin_amdgcn_wmma_f32_16x16x32_bf16(false, a1.v, false, b.v,
                                                        (short)0, acc1[t], false, false);
    }
  }
#pragma unroll
  for (int half = 0; half < 2; ++half) {
#pragma unroll
    for (int t = 0; t < 4; ++t) {
#pragma unroll
      for (int i = 0; i < 8; ++i) {
        int m = (lane >> 4) * 8 + i;
        int row = rowTile + half * 16 + m;
        int c = colBase + t * 16 + col;
        if (row < M && c < Ncol) {
          float v = half ? acc1[t][i] : acc0[t][i];
          size_t idx = (size_t)row * Ncol + c;
          if (addB) v += bf2f(addB[idx]);
          if (relu) v = fmaxf(v, 0.0f);
          if (outF) outF[idx] = v;
          if (outB) outB[idx] = f2bf(v);
        }
      }
    }
  }
}

// ---------------------------------------------------------------------------
// Flash-attention, one wave per (q-tile, head, bv*split). dh=16 padded to K=32.
// 32 keys/iteration: 2 WMMAs for S, online softmax, 1 WMMA for O.
// If part != null, writes unnormalized (O, m, l) partials for a split-K merge.
// ---------------------------------------------------------------------------
__global__ void flash_attn_kernel(const u16* __restrict__ Q, long qStride, int qPerB,
                                  const u16* __restrict__ Kp, long kStride, int kPerB,
                                  const u16* __restrict__ Vp, long vStride, int vPerB,
                                  u16* __restrict__ Out, long oStride,
                                  float* __restrict__ part,
                                  const int* __restrict__ mask,
                                  int N, int P, int nSplit, float scale)
{
  __shared__ float sP[16 * 32];
  int lane = threadIdx.x & 31;
  int h = blockIdx.y;
  int zz = blockIdx.z;
  int bv = zz / nSplit;
  int sp = zz - bv * nSplit;
  int b = bv / V_, vv = bv - b * V_;
  const u16* q = Q + (size_t)(qPerB ? b : bv) * qStride;
  const u16* k = Kp + (size_t)(kPerB ? b : bv) * kStride;
  const u16* v = Vp + (size_t)(vPerB ? b : bv) * vStride;

  int chunk = ((N + nSplit - 1) / nSplit + 31) & ~31;
  int n0 = sp * chunk;
  int n1 = n0 + chunk; if (n1 > N) n1 = N;

  int col = lane & 15;
  int half = lane >> 4;
  int khalf = half * 8;
  int qr = blockIdx.x * 16 + col;
  int qc = (qr < P) ? qr : (P - 1);

  Frag qa;
#pragma unroll
  for (int j = 0; j < 4; ++j)
    qa.u[j] = *(const u32*)(q + (size_t)qc * C_ + h * DH_ + khalf + 2 * j);
#pragma unroll
  for (int j = 0; j < 4; ++j) qa.u[4 + j] = 0;  // dh=16 pad

  v8f oacc = (v8f){0, 0, 0, 0, 0, 0, 0, 0};
  float mrow[8], lrow[8];
#pragma unroll
  for (int i = 0; i < 8; ++i) { mrow[i] = -1e30f; lrow[i] = 0.0f; }

  for (int nt = n0; nt < n1; nt += 32) {
    int k0 = nt + col, k1 = nt + 16 + col;
    int k0c = (k0 < N) ? k0 : (N - 1);
    int k1c = (k1 < N) ? k1 : (N - 1);
    Frag kb0, kb1;
#pragma unroll
    for (int j = 0; j < 4; ++j) {
      kb0.u[j] = *(const u32*)(k + (size_t)k0c * C_ + h * DH_ + khalf + 2 * j);
      kb1.u[j] = *(const u32*)(k + (size_t)k1c * C_ + h * DH_ + khalf + 2 * j);
      kb0.u[4 + j] = 0; kb1.u[4 + j] = 0;
    }
    v8f z = (v8f){0, 0, 0, 0, 0, 0, 0, 0};
    v8f s0 = __builtin_amdgcn_wmma_f32_16x16x32_bf16(false, qa.v, false, kb0.v, (short)0, z, false, false);
    v8f s1 = __builtin_amdgcn_wmma_f32_16x16x32_bf16(false, qa.v, false, kb1.v, (short)0, z, false, false);

    float bias0 = (k0 < n1) ? 0.0f : -1e30f;
    float bias1 = (k1 < n1) ? 0.0f : -1e30f;
    if (mask) {
      if (!(mask[b * P_ * V_ + k0c * V_ + vv] > 0)) bias0 = -1e30f;
      if (!(mask[b * P_ * V_ + k1c * V_ + vv] > 0)) bias1 = -1e30f;
    }
#pragma unroll
    for (int i = 0; i < 8; ++i) {
      float a0 = s0[i] * scale + bias0;
      float a1 = s1[i] * scale + bias1;
      float tm = fmaxf(a0, a1);
      for (int off = 8; off; off >>= 1) tm = fmaxf(tm, __shfl_xor(tm, off, 16));
      float nm = fmaxf(mrow[i], tm);
      float alpha = __expf(mrow[i] - nm);
      float p0 = __expf(a0 - nm);
      float p1 = __expf(a1 - nm);
      float ps = p0 + p1;
      for (int off = 8; off; off >>= 1) ps += __shfl_xor(ps, off, 16);
      mrow[i] = nm;
      lrow[i] = lrow[i] * alpha + ps;
      oacc[i] = oacc[i] * alpha;
      int m = half * 8 + i;
      sP[m * 32 + col] = p0;
      sP[m * 32 + col + 16] = p1;
    }
    __syncthreads();
    Frag pa, vb;
    int pr = col;  // A-fragment row = lane & 15
#pragma unroll
    for (int j = 0; j < 8; ++j) {
      int kk = (j < 4) ? (khalf + 2 * j) : (16 + khalf + 2 * (j - 4));
      u16 lo = f2bf(sP[pr * 32 + kk]);
      u16 hi = f2bf(sP[pr * 32 + kk + 1]);
      pa.u[j] = (u32)lo | ((u32)hi << 16);
      int g0 = nt + kk, g1 = nt + kk + 1;
      if (g0 >= N) g0 = N - 1;
      if (g1 >= N) g1 = N - 1;
      u16 vlo = v[(size_t)g0 * C_ + h * DH_ + col];
      u16 vhi = v[(size_t)g1 * C_ + h * DH_ + col];
      vb.u[j] = (u32)vlo | ((u32)vhi << 16);
    }
    oacc = __builtin_amdgcn_wmma_f32_16x16x32_bf16(false, pa.v, false, vb.v, (short)0, oacc, false, false);
    __syncthreads();
  }

  if (part) {
    // unnormalized partials: 16 rows x (16 O-cols + m + l)
    float* pb = part + (((size_t)zz * H_ + h) * gridDim.x + blockIdx.x) * (16 * 18);
#pragma unroll
    for (int i = 0; i < 8; ++i) {
      int m = half * 8 + i;
      pb[m * 18 + col] = oacc[i];
      if (col == 0) { pb[m * 18 + 16] = mrow[i]; pb[m * 18 + 17] = lrow[i]; }
    }
  } else {
    u16* o = Out + (size_t)bv * oStride;
#pragma unroll
    for (int i = 0; i < 8; ++i) {
      int m = half * 8 + i;
      int row = blockIdx.x * 16 + m;
      if (row < P) {
        float val = oacc[i] / (lrow[i] + 1e-20f);
        o[(size_t)row * C_ + h * DH_ + col] = f2bf(val);
      }
    }
  }
}

// Log-sum-exp merge of split-K attention partials.
__global__ void attn_merge_kernel(const float* __restrict__ part, u16* __restrict__ Out,
                                  long oStride, int nSplit, int qTiles, int P)
{
  int tid = threadIdx.x;
  int row = tid >> 4, col = tid & 15;
  int qt = blockIdx.x, h = blockIdx.y, bv = blockIdx.z;
  float M = -1e30f;
  for (int s = 0; s < nSplit; ++s) {
    const float* pb = part + (((size_t)(bv * nSplit + s) * H_ + h) * qTiles + qt) * (16 * 18);
    M = fmaxf(M, pb[row * 18 + 16]);
  }
  float L = 0.0f, O = 0.0f;
  for (int s = 0; s < nSplit; ++s) {
    const float* pb = part + (((size_t)(bv * nSplit + s) * H_ + h) * qTiles + qt) * (16 * 18);
    float w = __expf(pb[row * 18 + 16] - M);
    L += pb[row * 18 + 17] * w;
    O += pb[row * 18 + col] * w;
  }
  int q = qt * 16 + row;
  if (q < P)
    Out[(size_t)bv * oStride + (size_t)q * C_ + h * DH_ + col] = f2bf(O / (L + 1e-20f));
}

// ---------------------------------------------------------------------------
// Residual + LayerNorm (row per block, 128 threads)
// ---------------------------------------------------------------------------
__global__ void ln_kernel(const float* __restrict__ a, const float* __restrict__ res, int perB,
                          const float* __restrict__ g, const float* __restrict__ beta,
                          float* __restrict__ outF, u16* __restrict__ outB)
{
  __shared__ float red[C_];
  int row = blockIdx.x, c = threadIdx.x;
  size_t idx = (size_t)row * C_ + c;
  float x = a[idx];
  if (res) {
    int rr = perB ? ((row / (V_ * P_)) * P_ + (row % P_)) : row;
    x += res[(size_t)rr * C_ + c];
  }
  red[c] = x; __syncthreads();
  for (int s = 64; s; s >>= 1) { if (c < s) red[c] += red[c + s]; __syncthreads(); }
  float mean = red[0] / C_;
  __syncthreads();
  float d = x - mean;
  red[c] = d * d; __syncthreads();
  for (int s = 64; s; s >>= 1) { if (c < s) red[c] += red[c + s]; __syncthreads(); }
  float var = red[0] / C_;
  float y = d * rsqrtf(var + 1e-5f) * g[c] + beta[c];
  if (outF) outF[idx] = y;
  if (outB) outB[idx] = f2bf(y);
}

__global__ void add_to_bf16_kernel(const float* __restrict__ x, const float* __restrict__ pe,
                                   int perB, u16* __restrict__ out)
{
  int row = blockIdx.x, c = threadIdx.x;
  size_t idx = (size_t)row * C_ + c;
  int rr = perB ? ((row / (V_ * P_)) * P_ + (row % P_)) : row;
  out[idx] = f2bf(x[idx] + pe[(size_t)rr * C_ + c]);
}

__global__ void wt_kernel(const float* __restrict__ W, u16* __restrict__ Wt, int K, int Ncol)
{
  int i = blockIdx.x * blockDim.x + threadIdx.x;
  if (i >= K * Ncol) return;
  int k = i / Ncol, n = i - k * Ncol;
  Wt[(size_t)n * K + k] = f2bf(W[i]);
}

__global__ void queryT_kernel(const float* __restrict__ qf, float* __restrict__ outF,
                              u16* __restrict__ outB)
{
  int row = blockIdx.x;  // b*P+p
  int c = threadIdx.x;
  int b = row / P_, p = row - b * P_;
  float v = qf[(size_t)b * C_ * P_ + (size_t)c * P_ + p];
  outF[(size_t)row * C_ + c] = v;
  outB[(size_t)row * C_ + c] = f2bf(v);
}

__global__ void qpe_kernel(const float* __restrict__ qpos, const float* __restrict__ w1,
                           const float* __restrict__ b1, const float* __restrict__ w2,
                           const float* __restrict__ b2, float* __restrict__ outF,
                           u16* __restrict__ outB)
{
  __shared__ float hid[C_];
  int row = blockIdx.x, c = threadIdx.x;
  float q0 = qpos[row * 3 + 0], q1 = qpos[row * 3 + 1], q2 = qpos[row * 3 + 2];
  float hv = b1[c] + q0 * w1[c] + q1 * w1[C_ + c] + q2 * w1[2 * C_ + c];
  hid[c] = fmaxf(hv, 0.0f); __syncthreads();
  float o = b2[c];
  for (int d = 0; d < C_; ++d) o += hid[d] * w2[d * C_ + c];
  outF[(size_t)row * C_ + c] = o;
  outB[(size_t)row * C_ + c] = f2bf(o);
}

__global__ void prep_kernel(const float* __restrict__ l2i, const float* __restrict__ R,
                            const float* __restrict__ t, float* __restrict__ inv,
                            float* __restrict__ camt)
{
  int i = threadIdx.x;
  if (i >= BV_) return;
  float a[4][8];
  for (int r = 0; r < 4; ++r)
    for (int cc = 0; cc < 4; ++cc) {
      a[r][cc] = l2i[i * 16 + r * 4 + cc];
      a[r][4 + cc] = (r == cc) ? 1.0f : 0.0f;
    }
  for (int cp = 0; cp < 4; ++cp) {
    float ip = 1.0f / a[cp][cp];
    for (int j = 0; j < 8; ++j) a[cp][j] *= ip;
    for (int r = 0; r < 4; ++r) if (r != cp) {
      float f = a[r][cp];
      for (int j = 0; j < 8; ++j) a[r][j] -= f * a[cp][j];
    }
  }
  for (int r = 0; r < 4; ++r)
    for (int cc = 0; cc < 4; ++cc)
      inv[i * 16 + r * 4 + cc] = a[r][4 + cc];
  for (int d = 0; d < 3; ++d) {
    float s = 0.0f;
    for (int j = 0; j < 3; ++j) s += R[i * 9 + j * 3 + d] * t[i * 3 + j];
    camt[i * 3 + d] = -s;
  }
}

__global__ void kpos_hidden_kernel(const float* __restrict__ np, const float* __restrict__ inv,
                                   const float* __restrict__ camt, const float* __restrict__ w1,
                                   const float* __restrict__ b1, u16* __restrict__ kh)
{
  long blk = (long)blockIdx.x;  // bv*N + n
  int bv = (int)(blk / N_);
  int n = (int)(blk - (long)bv * N_);
  int c = threadIdx.x;
  float nx = np[n * 2 + 0], ny = np[n * 2 + 1];
  float px = 800.0f / (1.0f + __expf(-nx));
  float py = 448.0f / (1.0f + __expf(-ny));
  float p4[4] = {px, py, 1.0f, 1.0f};
  float kp[6];
  for (int r = 0; r < 3; ++r) {
    float s = 0.0f;
    for (int j = 0; j < 4; ++j) s += inv[bv * 16 + r * 4 + j] * p4[j];
    kp[r] = s;
  }
  for (int r = 0; r < 3; ++r) kp[3 + r] = camt[bv * 3 + r];
  float hv = b1[c];
  for (int j = 0; j < 6; ++j) hv += kp[j] * w1[j * C_ + c];
  kh[(size_t)blk * C_ + c] = f2bf(fmaxf(hv, 0.0f));
}

__global__ void keys_kernel(const float* __restrict__ l0, const float* __restrict__ l1,
                            u16* __restrict__ keys)
{
  long blk = (long)blockIdx.x;
  int bv = (int)(blk / N_);
  int n = (int)(blk - (long)bv * N_);
  int c = threadIdx.x;
  float f = (n < 5600)
      ? l0[((size_t)bv * C_ + c) * 5600 + n]
      : l1[((size_t)bv * C_ + c) * 1400 + (n - 5600)];
  keys[(size_t)blk * C_ + c] = f2bf(f);
}

__global__ void valid_kernel(const int* __restrict__ mask, int* __restrict__ valid)
{
  __shared__ int red[256];
  int bv = blockIdx.x;
  int b = bv / V_, vv = bv - b * V_;
  int tid = threadIdx.x;
  int s = 0;
  for (int p = tid; p < P_; p += 256) s += (mask[b * P_ * V_ + p * V_ + vv] > 0) ? 1 : 0;
  red[tid] = s; __syncthreads();
  for (int st = 128; st; st >>= 1) { if (tid < st) red[tid] += red[tid + st]; __syncthreads(); }
  if (tid == 0) valid[bv] = (red[0] > 1) ? 1 : 0;
}

__global__ void avg_kernel(const float* __restrict__ outL, const int* __restrict__ mask,
                           const int* __restrict__ valid, float* __restrict__ qfeat)
{
  int row = blockIdx.x;  // b*P + p
  int b = row / P_, p = row - b * P_;
  int c = threadIdx.x;
  float s = 0.0f, cnt = 0.0f;
  for (int vv = 0; vv < V_; ++vv) {
    float w = ((mask[b * P_ * V_ + p * V_ + vv] > 0) && valid[b * V_ + vv]) ? 1.0f : 0.0f;
    s += w * outL[((size_t)(b * V_ + vv) * P_ + p) * C_ + c];
    cnt += w;
  }
  qfeat[(size_t)b * C_ * P_ + (size_t)c * P_ + p] = s / (cnt + 1e-4f);
}

__global__ void head_kernel(const float* __restrict__ qfeat, const float* __restrict__ qpos_in,
                            const float* __restrict__ w1, const float* __restrict__ b1,
                            const float* __restrict__ w2, const float* __restrict__ b2,
                            float* __restrict__ qpos_out, float* __restrict__ bevT,
                            float* __restrict__ bev)
{
  __shared__ float qf[C_];
  __shared__ float h1[C_];
  int row = blockIdx.x;  // b*P+p
  int b = row / P_, p = row - b * P_;
  int c = threadIdx.x;
  qf[c] = qfeat[(size_t)b * C_ * P_ + (size_t)c * P_ + p];
  __syncthreads();
  float acc = b1[c];
  for (int d = 0; d < C_; ++d) acc += qf[d] * w1[d * C_ + c];
  h1[c] = fmaxf(acc, 0.0f);
  __syncthreads();
  if (c < 3) {
    float o = b2[c];
    for (int d = 0; d < C_; ++d) o += h1[d] * w2[d * 3 + c];
    o += qpos_in[(size_t)row * 3 + c];
    qpos_out[(size_t)row * 3 + c] = o;
    if (c < 2) {
      float bvv = (o + 54.0f) / 108.0f * 180.0f;
      bevT[(size_t)row * 2 + c] = bvv;
      bev[(size_t)b * 2 * P_ + (size_t)c * P_ + p] = bvv;
    }
  }
}

// ---------------------------------------------------------------------------
extern "C" void kernel_launch(void* const* d_in, const int* in_sizes, int n_in,
                              void* d_out, int out_size, void* d_ws, size_t ws_size,
                              hipStream_t stream)
{
  (void)in_sizes; (void)n_in; (void)out_size; (void)ws_size;
  const float* img_query_feat = (const float*)d_in[0];
  const float* img_query_pos  = (const float*)d_in[1];
  const int*   view_mask      = (const int*)d_in[2];
  const float* feat_l0        = (const float*)d_in[3];
  const float* feat_l1        = (const float*)d_in[4];
  const float* normal_pos     = (const float*)d_in[5];
  const float* l2i            = (const float*)d_in[6];
  const float* l2c_r          = (const float*)d_in[7];
  const float* l2c_t          = (const float*)d_in[8];
  const float* h1_b  = (const float*)d_in[9];
  const float* h1_w  = (const float*)d_in[10];
  const float* h2_b  = (const float*)d_in[11];
  const float* h2_w  = (const float*)d_in[12];
  const float* ca_k_b = (const float*)d_in[13];
  const float* ca_k_w = (const float*)d_in[14];
  const float* ca_o_b = (const float*)d_in[15];
  const float* ca_o_w = (const float*)d_in[16];
  const float* ca_q_b = (const float*)d_in[17];
  const float* ca_q_w = (const float*)d_in[18];
  const float* ca_v_b = (const float*)d_in[19];
  const float* ca_v_w = (const float*)d_in[20];
  const float* cp0_b = (const float*)d_in[21];
  const float* cp0_w = (const float*)d_in[22];
  const float* cp1_b = (const float*)d_in[23];
  const float* cp1_w = (const float*)d_in[24];
  const float* ff1_b = (const float*)d_in[25];
  const float* ff1_w = (const float*)d_in[26];
  const float* ff2_b = (const float*)d_in[27];
  const float* ff2_w = (const float*)d_in[28];
  const float* ln1_b = (const float*)d_in[29];
  const float* ln1_g = (const float*)d_in[30];
  const float* ln2_b = (const float*)d_in[31];
  const float* ln2_g = (const float*)d_in[32];
  const float* ln3_b = (const float*)d_in[33];
  const float* ln3_g = (const float*)d_in[34];
  const float* sa_k_b = (const float*)d_in[35];
  const float* sa_k_w = (const float*)d_in[36];
  const float* sa_o_b = (const float*)d_in[37];
  const float* sa_o_w = (const float*)d_in[38];
  const float* sa_q_b = (const float*)d_in[39];
  const float* sa_q_w = (const float*)d_in[40];
  const float* sa_v_b = (const float*)d_in[41];
  const float* sa_v_w = (const float*)d_in[42];
  const float* sp0_b = (const float*)d_in[43];
  const float* sp0_w = (const float*)d_in[44];
  const float* sp1_b = (const float*)d_in[45];
  const float* sp1_w = (const float*)d_in[46];

  char* ws = (char*)d_ws;
  size_t off = 0;
  auto alloc = [&](size_t bytes) -> void* {
    void* p = ws + off;
    off = (off + bytes + 255) & ~((size_t)255);
    return p;
  };

  float* invM = (float*)alloc(BV_ * 16 * 4);
  float* camt = (float*)alloc(BV_ * 3 * 4);
  int*   valid = (int*)alloc(BV_ * 4);
  u16* sa_q_t = (u16*)alloc(C_ * C_ * 2);
  u16* sa_k_t = (u16*)alloc(C_ * C_ * 2);
  u16* sa_v_t = (u16*)alloc(C_ * C_ * 2);
  u16* sa_o_t = (u16*)alloc(C_ * C_ * 2);
  u16* ca_q_t = (u16*)alloc(C_ * C_ * 2);
  u16* ca_k_t = (u16*)alloc(C_ * C_ * 2);
  u16* ca_v_t = (u16*)alloc(C_ * C_ * 2);
  u16* ca_o_t = (u16*)alloc(C_ * C_ * 2);
  u16* cp1_t  = (u16*)alloc(C_ * C_ * 2);
  u16* ff1_t  = (u16*)alloc((size_t)FFN_ * C_ * 2);
  u16* ff2_t  = (u16*)alloc((size_t)C_ * FFN_ * 2);
  float* queryF = (float*)alloc((size_t)B_ * P_ * C_ * 4);
  u16*   queryB = (u16*)alloc((size_t)B_ * P_ * C_ * 2);
  float* qpeF   = (float*)alloc((size_t)B_ * P_ * C_ * 4);
  u16*   qpeB   = (u16*)alloc((size_t)B_ * P_ * C_ * 2);
  u16*   qsaB   = (u16*)alloc((size_t)B_ * P_ * C_ * 2);
  u16*   saQ    = (u16*)alloc((size_t)B_ * P_ * C_ * 2);
  u16*   saK    = (u16*)alloc((size_t)B_ * P_ * C_ * 2);
  u16*   saV    = (u16*)alloc((size_t)B_ * P_ * C_ * 2);
  u16*   saAttn = (u16*)alloc((size_t)BV_ * P_ * C_ * 2);
  float* saO    = (float*)alloc((size_t)BV_ * P_ * C_ * 4);
  float* x1     = (float*)alloc((size_t)BV_ * P_ * C_ * 4);
  u16*   xq     = (u16*)alloc((size_t)BV_ * P_ * C_ * 2);
  u16*   Qc     = (u16*)alloc((size_t)BV_ * P_ * C_ * 2);
  u16*   caAttn = (u16*)alloc((size_t)BV_ * P_ * C_ * 2);
  float* caO    = (float*)alloc((size_t)BV_ * P_ * C_ * 4);
  float* x2     = (float*)alloc((size_t)BV_ * P_ * C_ * 4);
  u16*   x2b    = (u16*)alloc((size_t)BV_ * P_ * C_ * 2);
  u16*   ffh    = (u16*)alloc((size_t)BV_ * P_ * FFN_ * 2);
  float* ffo    = (float*)alloc((size_t)BV_ * P_ * C_ * 4);
  float* outL   = (float*)alloc((size_t)BV_ * P_ * C_ * 4);
  float* attnPart = (float*)alloc((size_t)BV_ * SPLIT_ * H_ * QT_ * 16 * 18 * 4);
  u16* kh    = (u16*)alloc((size_t)BV_ * N_ * C_ * 2);
  u16* keysB = (u16*)alloc((size_t)BV_ * N_ * C_ * 2);
  u16* kin   = (u16*)alloc((size_t)BV_ * N_ * C_ * 2);
  u16* Kproj = kh;   // reuse: kh consumed by kpe GEMM
  u16* Vproj = kin;  // reuse: kin consumed by K-proj GEMM

  prep_kernel<<<1, 32, 0, stream>>>(l2i, l2c_r, l2c_t, invM, camt);

  auto wt = [&](const float* W, u16* Wt, int K, int Nc) {
    int tot = K * Nc;
    wt_kernel<<<(tot + 255) / 256, 256, 0, stream>>>(W, Wt, K, Nc);
  };
  wt(sa_q_w, sa_q_t, C_, C_); wt(sa_k_w, sa_k_t, C_, C_);
  wt(sa_v_w, sa_v_t, C_, C_); wt(sa_o_w, sa_o_t, C_, C_);
  wt(ca_q_w, ca_q_t, C_, C_); wt(ca_k_w, ca_k_t, C_, C_);
  wt(ca_v_w, ca_v_t, C_, C_); wt(ca_o_w, ca_o_t, C_, C_);
  wt(cp1_w, cp1_t, C_, C_);
  wt(ff1_w, ff1_t, C_, FFN_);
  wt(ff2_w, ff2_t, FFN_, C_);

  queryT_kernel<<<B_ * P_, C_, 0, stream>>>(img_query_feat, queryF, queryB);
  qpe_kernel<<<B_ * P_, C_, 0, stream>>>(img_query_pos, sp0_w, sp0_b, sp1_w, sp1_b, qpeF, qpeB);
  add_to_bf16_kernel<<<B_ * P_, C_, 0, stream>>>(queryF, qpeF, 0, qsaB);
  kpos_hidden_kernel<<<BV_ * N_, C_, 0, stream>>>(normal_pos, invM, camt, cp0_w, cp0_b, kh);
  keys_kernel<<<BV_ * N_, C_, 0, stream>>>(feat_l0, feat_l1, keysB);

  auto gemm = [&](const u16* A, const u16* W, const float* bias, const u16* addB,
                  float* oF, u16* oB, int M, int Nc, int K, int relu) {
    dim3 g(Nc / 64, (M + 127) / 128);
    gemm_bf16_kernel<<<g, 128, 0, stream>>>(A, W, bias, addB, oF, oB, M, Nc, K, relu);
  };

  // kpe = hidden @ W + b ; kin = key + kpe
  gemm(kh, cp1_t, cp1_b, keysB, nullptr, kin, BV_ * N_, C_, C_, 0);
  // K and V projections for cross-attention
  gemm(kin, ca_k_t, ca_k_b, nullptr, nullptr, Kproj, BV_ * N_, C_, C_, 0);
  gemm(keysB, ca_v_t, ca_v_b, nullptr, nullptr, Vproj, BV_ * N_, C_, C_, 0);

  // self-attention projections (per batch, shared over views)
  gemm(qsaB, sa_q_t, sa_q_b, nullptr, nullptr, saQ, B_ * P_, C_, C_, 0);
  gemm(qsaB, sa_k_t, sa_k_b, nullptr, nullptr, saK, B_ * P_, C_, C_, 0);
  gemm(queryB, sa_v_t, sa_v_b, nullptr, nullptr, saV, B_ * P_, C_, C_, 0);
  {
    dim3 g(QT_, H_, BV_);  // single split, masked, final store
    flash_attn_kernel<<<g, 32, 0, stream>>>(saQ, (long)P_ * C_, 1, saK, (long)P_ * C_, 1,
                                            saV, (long)P_ * C_, 1, saAttn, (long)P_ * C_,
                                            nullptr, view_mask, P_, P_, 1, 0.25f);
  }
  gemm(saAttn, sa_o_t, sa_o_b, nullptr, saO, nullptr, BV_ * P_, C_, C_, 0);
  ln_kernel<<<BV_ * P_, C_, 0, stream>>>(saO, queryF, 1, ln1_g, ln1_b, x1, nullptr);

  add_to_bf16_kernel<<<BV_ * P_, C_, 0, stream>>>(x1, qpeF, 1, xq);
  gemm(xq, ca_q_t, ca_q_b, nullptr, nullptr, Qc, BV_ * P_, C_, C_, 0);
  {
    dim3 g(QT_, H_, BV_ * SPLIT_);  // split-K over the 7000 keys
    flash_attn_kernel<<<g, 32, 0, stream>>>(Qc, (long)P_ * C_, 0, Kproj, (long)N_ * C_, 0,
                                            Vproj, (long)N_ * C_, 0, caAttn, (long)P_ * C_,
                                            attnPart, nullptr, N_, P_, SPLIT_, 0.25f);
    dim3 gm(QT_, H_, BV_);
    attn_merge_kernel<<<gm, 256, 0, stream>>>(attnPart, caAttn, (long)P_ * C_,
                                              SPLIT_, QT_, P_);
  }
  gemm(caAttn, ca_o_t, ca_o_b, nullptr, caO, nullptr, BV_ * P_, C_, C_, 0);
  ln_kernel<<<BV_ * P_, C_, 0, stream>>>(caO, x1, 0, ln2_g, ln2_b, x2, x2b);

  gemm(x2b, ff1_t, ff1_b, nullptr, nullptr, ffh, BV_ * P_, FFN_, C_, 1);
  gemm(ffh, ff2_t, ff2_b, nullptr, ffo, nullptr, BV_ * P_, C_, FFN_, 0);
  ln_kernel<<<BV_ * P_, C_, 0, stream>>>(ffo, x2, 0, ln3_g, ln3_b, outL, nullptr);

  valid_kernel<<<BV_, 256, 0, stream>>>(view_mask, valid);
  float* qfeat_out = (float*)d_out;
  float* qpos_out  = qfeat_out + (size_t)B_ * C_ * P_;
  float* bevT_out  = qpos_out + (size_t)B_ * P_ * 3;
  float* bev_out   = bevT_out + (size_t)B_ * P_ * 2;
  avg_kernel<<<B_ * P_, C_, 0, stream>>>(outL, view_mask, valid, qfeat_out);
  head_kernel<<<B_ * P_, C_, 0, stream>>>(qfeat_out, img_query_pos, h1_w, h1_b, h2_w, h2_b,
                                          qpos_out, bevT_out, bev_out);
}
